// CANA_nei_1_70437463654947
// MI455X (gfx1250) — compile-verified, matrix operand order checked by reading, not run on hardware
//
#include <hip/hip_runtime.h>
#include <hip/hip_bf16.h>
#include <math.h>

#define IN_DIM 512
#define HID    128
#define LAYERS 2

typedef __attribute__((ext_vector_type(16))) _Float16 v16h;
typedef __attribute__((ext_vector_type(8)))  _Float16 v8h;
typedef __attribute__((ext_vector_type(8)))  float    v8f;

// ---------------------------------------------------------------------------
// WMMA helpers (CDNA5 gfx1250, wave32). Layouts per ISA 7.12.2:
//   A 16x32 f16: lane L<16 -> M=L, K = {0..7, 16..23}; lane>=16 -> M=L-16,
//                K = {8..15, 24..31}
//   B 32x16 f16: mirrored, lane selects output column N, same K striping.
//   C 16x16 f32: VGPR r -> M = r + (lane>=16 ? 8 : 0), N = lane&15.
// ---------------------------------------------------------------------------
__device__ __forceinline__ v8f wmma_f16(v16h a, v16h b, v8f c) {
    return __builtin_amdgcn_wmma_f32_16x16x32_f16(
        /*neg_a=*/false, a, /*neg_b=*/false, b,
        /*c_mod=*/(short)0, c, /*reuse_a=*/false, /*reuse_b=*/false);
}

// A fragment from contiguous f32 memory (conversion path; used for fea/agg).
__device__ __forceinline__ v16h load_a_f32(const float* base, int lane) {
    v16h a;
    const int koff = (lane & 16) ? 8 : 0;
    const float* p = base + koff;
#pragma unroll
    for (int i = 0; i < 8; ++i) a[i]     = (_Float16)p[i];
#pragma unroll
    for (int i = 0; i < 8; ++i) a[i + 8] = (_Float16)p[i + 16];
    return a;
}

// A fragment from contiguous f16 memory: base points at row start + kb
// (kb multiple of 32 -> 64B offset). Both 16B chunks are 16B-aligned.
__device__ __forceinline__ v16h load_a_f16(const _Float16* base, int lane) {
    const _Float16* p = base + ((lane & 16) ? 8 : 0);
    const v8h lo = *(const v8h*)p;
    const v8h hi = *(const v8h*)(p + 16);
    v16h a;
#pragma unroll
    for (int i = 0; i < 8; ++i) { a[i] = lo[i]; a[i + 8] = hi[i]; }
    return a;
}

// Packed-B layout: record (tile, n) = 32 consecutive halves (64B):
//   halves [0..15]  = lane<16  pattern, halves [16..31] = lane>=16 pattern.
// Each lane reads one contiguous, 32-byte-aligned 32B chunk -> 2x b128.
__device__ __forceinline__ v16h load_b_packed(const _Float16* __restrict__ P,
                                              int tile, int ncol, int npad, int lane) {
    const _Float16* p = P + ((size_t)tile * npad + ncol) * 32 + ((lane & 16) ? 16 : 0);
    return *(const v16h*)p;
}

__device__ __forceinline__ float gelu_exact(float v) {
    return 0.5f * v * (1.0f + erff(v * 0.70710678118654752f));
}

// ---------------------------------------------------------------------------
// Pack f32 weight W[K x Nout] (row-major) into WMMA B-fragment f16 records.
// Columns >= Nout (up to NoutPad) are zero-filled. K must be multiple of 32.
// ---------------------------------------------------------------------------
__global__ void pack_b_kernel(const float* __restrict__ W, _Float16* __restrict__ dst,
                              int K, int Nout, int NoutPad) {
    const int idx = blockIdx.x * 256 + threadIdx.x;
    const int total = (K / 32) * NoutPad * 32;
    if (idx >= total) return;
    const int slot = idx & 31;
    const int rec  = idx >> 5;
    const int n = rec % NoutPad;
    const int t = rec / NoutPad;
    const int koff = (slot & 16) ? 8 : 0;
    const int i = slot & 15;
    const int k = t * 32 + koff + (i < 8 ? i : 16 + (i - 8));
    const float v = (n < Nout) ? W[(size_t)k * Nout + n] : 0.0f;
    dst[idx] = (_Float16)v;
}

// ---------------------------------------------------------------------------
// fx: out = gelu(gelu(fea @ W1 + b1) @ W2 + b2); also writes f16 mirror outh.
// Block = 256 threads (8 waves). Block owns 16 rows; wave w owns cols 16w..16w+15.
// ---------------------------------------------------------------------------
__global__ void fx_kernel(const float* __restrict__ fea,
                          const _Float16* __restrict__ W1f, const float* __restrict__ b1,
                          const _Float16* __restrict__ W2f, const float* __restrict__ b2,
                          float* __restrict__ out, _Float16* __restrict__ outh, int Nn) {
    __shared__ _Float16 hbuf[16 * HID];       // layer-1 activations, f16
    const int lane = threadIdx.x & 31;
    const int wave = threadIdx.x >> 5;
    const int mb   = blockIdx.x * 16;
    const int n    = wave * 16 + (lane & 15);
    int mrow = mb + (lane & 15);
    if (mrow >= Nn) mrow = Nn - 1;

    // ---- layer 1: [16 x 512] @ [512 x 128] tile ----
    v8f c = {};
#pragma unroll 4
    for (int kb = 0; kb < IN_DIM; kb += 32) {
        v16h a = load_a_f32(fea + (size_t)mrow * IN_DIM + kb, lane);
        v16h b = load_b_packed(W1f, kb >> 5, n, HID, lane);
        c = wmma_f16(a, b, c);
    }
    const int mtop = (lane & 16) ? 8 : 0;
    const float bias1 = b1[n];
#pragma unroll
    for (int r = 0; r < 8; ++r)
        hbuf[(mtop + r) * HID + n] = (_Float16)gelu_exact(c[r] + bias1);
    __syncthreads();

    // ---- layer 2: [16 x 128] @ [128 x 128] tile ----
    v8f c2 = {};
#pragma unroll
    for (int kb = 0; kb < HID; kb += 32) {
        v16h a = load_a_f16(hbuf + (size_t)(lane & 15) * HID + kb, lane);
        v16h b = load_b_packed(W2f, kb >> 5, n, HID, lane);
        c2 = wmma_f16(a, b, c2);
    }
    const float bias2 = b2[n];
#pragma unroll
    for (int r = 0; r < 8; ++r) {
        const int m = mb + mtop + r;
        float v = gelu_exact(c2[r] + bias2);
        if (m < Nn) {
            out[(size_t)m * HID + n]  = v;
            outh[(size_t)m * HID + n] = (_Float16)v;
        }
    }
}

// ---------------------------------------------------------------------------
// edge attention: att[e] = sigmoid(relu(relu(relu(cat(x[row],x[col]) @ W1+b1)
//                                   @ W2+b2) @ W3+b3)); x read as f16 mirror.
// Block = 256 threads (8 waves); wave handles 16 edges.
// ---------------------------------------------------------------------------
__global__ void edge_att_kernel(const _Float16* __restrict__ xh,
                                const int* __restrict__ row, const int* __restrict__ col,
                                const _Float16* __restrict__ W1f, const float* __restrict__ b1,
                                const float* __restrict__ W2, const float* __restrict__ b2,
                                const float* __restrict__ W3, const float* __restrict__ b3,
                                float* __restrict__ att, int Ee) {
    __shared__ float h1s[8][16][20];
    const int lane = threadIdx.x & 31;
    const int wave = threadIdx.x >> 5;
    int e = blockIdx.x * 128 + wave * 16 + (lane & 15);
    if (e >= Ee) e = Ee - 1;
    const _Float16* xr = xh + (size_t)row[e] * HID;
    const _Float16* xc = xh + (size_t)col[e] * HID;
    const int n0 = lane & 15;

    v8f c0 = {}, c1 = {};
#pragma unroll
    for (int kb = 0; kb < 2 * HID; kb += 32) {
        const _Float16* base = (kb < HID) ? (xr + kb) : (xc + (kb - HID));
        v16h a   = load_a_f16(base, lane);
        v16h bf0 = load_b_packed(W1f, kb >> 5, n0,      32, lane);
        v16h bf1 = load_b_packed(W1f, kb >> 5, 16 + n0, 32, lane);
        c0 = wmma_f16(a, bf0, c0);
        c1 = wmma_f16(a, bf1, c1);
    }
    const int mtop = (lane & 16) ? 8 : 0;
    const float bb0 = b1[n0];
#pragma unroll
    for (int r = 0; r < 8; ++r)
        h1s[wave][mtop + r][n0] = fmaxf(c0[r] + bb0, 0.0f);
    if (n0 < 4) {
        const float bb1 = b1[16 + n0];
#pragma unroll
        for (int r = 0; r < 8; ++r)
            h1s[wave][mtop + r][16 + n0] = fmaxf(c1[r] + bb1, 0.0f);
    }
    __syncthreads();

    const int t = threadIdx.x;
    if (t < 128) {
        const int eg = blockIdx.x * 128 + t;
        if (eg < Ee) {
            const float* h1 = &h1s[t >> 4][t & 15][0];
            float h2[10];
#pragma unroll
            for (int j = 0; j < 10; ++j) {
                float acc = b2[j];
#pragma unroll
                for (int k = 0; k < 20; ++k) acc = fmaf(h1[k], W2[k * 10 + j], acc);
                h2[j] = fmaxf(acc, 0.0f);
            }
            float acc3 = b3[0];
#pragma unroll
            for (int j = 0; j < 10; ++j) acc3 = fmaf(h2[j], W3[j], acc3);
            acc3 = fmaxf(acc3, 0.0f);
            att[eg] = 1.0f / (1.0f + expf(-acc3));
        }
    }
}

// ---------------------------------------------------------------------------
// scatter: agg[col[e], :] += w(e) * xh[row[e], :]   (f16 gather, f32 atomics)
//   mode 0: w = 0.5*att (wc, hard==0)       mode 1: w = 1 - 0.5*att (wo)
// 256-thread block handles 2 edges; channel-per-lane -> coalesced.
// ---------------------------------------------------------------------------
__global__ void scatter_kernel(const _Float16* __restrict__ xh,
                               const int* __restrict__ row, const int* __restrict__ col,
                               const float* __restrict__ att,
                               float* __restrict__ agg, int Ee, int mode) {
    const int e = blockIdx.x * 2 + (threadIdx.x >> 7);
    if (e >= Ee) return;
    const int ch = threadIdx.x & 127;
    float w = 0.5f * att[e];
    if (mode) w = 1.0f - w;
    const int r = row[e], c = col[e];
    const float xv = (float)xh[(size_t)r * HID + ch];
    atomicAdd(&agg[(size_t)c * HID + ch], w * xv);
}

// ---------------------------------------------------------------------------
// conv: out = softsign(xh @ Wroot + agg @ Wrel); optional f16 mirror outh.
// Block = 256 threads (8 waves); block owns 16 rows, wave owns 16 cols.
// ---------------------------------------------------------------------------
__global__ void conv_kernel(const _Float16* __restrict__ xh, const float* __restrict__ agg,
                            const _Float16* __restrict__ Wrootf,
                            const _Float16* __restrict__ Wrelf,
                            float* __restrict__ out, _Float16* __restrict__ outh, int Nn) {
    const int lane = threadIdx.x & 31;
    const int wave = threadIdx.x >> 5;
    const int mb = blockIdx.x * 16;
    const int n  = wave * 16 + (lane & 15);
    int mrow = mb + (lane & 15);
    if (mrow >= Nn) mrow = Nn - 1;

    v8f c = {};
#pragma unroll
    for (int kb = 0; kb < HID; kb += 32) {
        v16h a = load_a_f16(xh + (size_t)mrow * HID + kb, lane);
        v16h b = load_b_packed(Wrootf, kb >> 5, n, HID, lane);
        c = wmma_f16(a, b, c);
    }
#pragma unroll
    for (int kb = 0; kb < HID; kb += 32) {
        v16h a = load_a_f32(agg + (size_t)mrow * HID + kb, lane);
        v16h b = load_b_packed(Wrelf, kb >> 5, n, HID, lane);
        c = wmma_f16(a, b, c);
    }
    const int mtop = (lane & 16) ? 8 : 0;
#pragma unroll
    for (int r = 0; r < 8; ++r) {
        const int m = mb + mtop + r;
        float v = c[r];
        v = v / (1.0f + fabsf(v));            // soft_sign
        if (m < Nn) {
            out[(size_t)m * HID + n] = v;
            if (outh) outh[(size_t)m * HID + n] = (_Float16)v;
        }
    }
}

// ---------------------------------------------------------------------------
// Host orchestration.
// Output stack order: [src1, xs_s0, xs_s1, trg1, xs_t0, xs_t1,
//                      cf_s0, cf_s1, cf_t0, cf_t1]  each [N,128].
// ---------------------------------------------------------------------------
extern "C" void kernel_launch(void* const* d_in, const int* in_sizes, int n_in,
                              void* d_out, int out_size, void* d_ws, size_t ws_size,
                              hipStream_t stream) {
    const float* src_fea = (const float*)d_in[0];
    const float* trg_fea = (const float*)d_in[1];
    const int*   src_row = (const int*)d_in[2];
    const int*   src_col = (const int*)d_in[3];
    const int*   trg_row = (const int*)d_in[4];
    const int*   trg_col = (const int*)d_in[5];
    const float* fxW1 = (const float*)d_in[6];
    const float* fxb1 = (const float*)d_in[7];
    const float* fxW2 = (const float*)d_in[8];
    const float* fxb2 = (const float*)d_in[9];
    const float* eaW1 = (const float*)d_in[10];
    const float* eab1 = (const float*)d_in[11];
    const float* eaW2 = (const float*)d_in[12];
    const float* eab2 = (const float*)d_in[13];
    const float* eaW3 = (const float*)d_in[14];
    const float* eab3 = (const float*)d_in[15];
    const float* convWroot = (const float*)d_in[16];
    const float* convWrel  = (const float*)d_in[17];

    const int Nn = in_sizes[0] / IN_DIM;      // 50000
    const int Ee = in_sizes[2];               // 600000
    float* out = (float*)d_out;
    const size_t slot = (size_t)Nn * HID;

    // ---- workspace layout ----
    char* ws = (char*)d_ws;
    size_t off = 0;
    auto take = [&](size_t bytes) { char* p = ws + off; off = (off + bytes + 255) & ~(size_t)255; return p; };
    float*    att    = (float*)take((size_t)Ee * 4);
    float*    agg    = (float*)take((size_t)Nn * HID * 4);
    _Float16* xhA    = (_Float16*)take((size_t)Nn * HID * 2);    // f16 activation mirrors
    _Float16* xhB    = (_Float16*)take((size_t)Nn * HID * 2);
    _Float16* fxW1f  = (_Float16*)take((size_t)(IN_DIM/32) * HID * 32 * 2);
    _Float16* fxW2f  = (_Float16*)take((size_t)(HID/32)    * HID * 32 * 2);
    _Float16* eaW1f  = (_Float16*)take((size_t)(256/32)    * 32  * 32 * 2);
    _Float16* cRootf = (_Float16*)take((size_t)(LAYERS*HID/32) * HID * 32 * 2);
    _Float16* cRelf  = (_Float16*)take((size_t)(LAYERS*HID/32) * HID * 32 * 2);
    const size_t aggBytes = (size_t)Nn * HID * 4;

    // ---- pack weights to WMMA B-fragment f16 layout ----
    auto packGrid = [](int K, int npad) { return ((K / 32) * npad * 32 + 255) / 256; };
    pack_b_kernel<<<packGrid(IN_DIM, HID), 256, 0, stream>>>(fxW1, fxW1f, IN_DIM, HID, HID);
    pack_b_kernel<<<packGrid(HID, HID),   256, 0, stream>>>(fxW2, fxW2f, HID, HID, HID);
    pack_b_kernel<<<packGrid(256, 32),    256, 0, stream>>>(eaW1, eaW1f, 256, 20, 32);
    pack_b_kernel<<<packGrid(LAYERS*HID, HID), 256, 0, stream>>>(convWroot, cRootf, LAYERS*HID, HID, HID);
    pack_b_kernel<<<packGrid(LAYERS*HID, HID), 256, 0, stream>>>(convWrel,  cRelf,  LAYERS*HID, HID, HID);

    const int gemmGrid = (Nn + 15) / 16;
    const int eaGrid   = (Ee + 127) / 128;
    const int scGrid   = (Ee + 1) / 2;

    for (int b = 0; b < 2; ++b) {
        const int* rowp = b ? trg_row : src_row;
        const int* colp = b ? trg_col : src_col;
        const float* fea = b ? trg_fea : src_fea;

        // fx -> x0 (f32 to its output slot) + f16 mirror in xhA
        fx_kernel<<<gemmGrid, 256, 0, stream>>>(fea, fxW1f, fxb1, fxW2f, fxb2,
                                                out + (b ? 3 : 0) * slot, xhA, Nn);
        _Float16* xcur = xhA;
        _Float16* xnxt = xhB;

        for (int i = 0; i < LAYERS; ++i) {
            const _Float16* Wrootf = cRootf + (size_t)i * (HID / 32) * HID * 32;
            const _Float16* Wrelf  = cRelf  + (size_t)i * (HID / 32) * HID * 32;

            // attention from the pre-update x
            edge_att_kernel<<<eaGrid, 256, 0, stream>>>(
                xcur, rowp, colp, eaW1f, eab1, eaW2, eab2, eaW3, eab3, att, Ee);

            // x_new = softsign(conv(x, wc)) -> f32 slot + f16 mirror
            hipMemsetAsync(agg, 0, aggBytes, stream);
            scatter_kernel<<<scGrid, 256, 0, stream>>>(xcur, rowp, colp, att, agg, Ee, 0);
            float* xnew = out + ((b ? 4 : 1) + i) * slot;
            conv_kernel<<<gemmGrid, 256, 0, stream>>>(xcur, agg, Wrootf, Wrelf, xnew, xnxt, Nn);

            // cfd = softsign(conv(x_new, wo))   (same att, new x; no mirror needed)
            hipMemsetAsync(agg, 0, aggBytes, stream);
            scatter_kernel<<<scGrid, 256, 0, stream>>>(xnxt, rowp, colp, att, agg, Ee, 1);
            float* cf = out + ((b ? 8 : 6) + i) * slot;
            conv_kernel<<<gemmGrid, 256, 0, stream>>>(xnxt, agg, Wrootf, Wrelf, cf, nullptr, Nn);

            _Float16* tmp = xcur; xcur = xnxt; xnxt = tmp;
        }
    }
}
